// VGAEModel_70274254897510
// MI455X (gfx1250) — matrix operand order repaired and checked by reading, not compiled, not run on hardware
//
#include <hip/hip_runtime.h>
#include <hip/hip_bf16.h>

// ---------------------------------------------------------------------------
// VGAE on gfx1250: bf16 WMMA GEMMs (pre-transposed B for b128 fragment loads),
// f32 atomic graph scatter, LDS-staged fused-sigmoid decoder sigmoid(z@z^T).
// Decoder is store-bound: 16384^2 * 4B = 1.07 GB => ~46us floor @ 23.3 TB/s.
// Decoder stores are non-temporal: write-once stream must not evict the hot
// 2 MB z panel from L2.
// ---------------------------------------------------------------------------

typedef __bf16 bf16_t;
typedef bf16_t v16bf __attribute__((ext_vector_type(16)));
typedef bf16_t v8bf  __attribute__((ext_vector_type(8)));
typedef float  v8f   __attribute__((ext_vector_type(8)));

#define GN 16384   // nodes
#define GE 262144  // edges

static __device__ __forceinline__ bf16_t f2bf(float f) {
  unsigned u = __float_as_uint(f);
  unsigned r = (u + 0x7FFFu + ((u >> 16) & 1u)) >> 16;   // round-to-nearest-even
  unsigned short s = (unsigned short)r;
  return __builtin_bit_cast(bf16_t, s);
}

static __device__ __forceinline__ v16bf mk16(v8bf lo, v8bf hi) {
  return __builtin_shufflevector(lo, hi, 0, 1, 2, 3, 4, 5, 6, 7,
                                 8, 9, 10, 11, 12, 13, 14, 15);
}

// ---------------- utility kernels -----------------------------------------

__global__ void zero_f32(float* __restrict__ p, long n) {
  long i = (long)blockIdx.x * blockDim.x + threadIdx.x;
  if (i < n) p[i] = 0.0f;
}

__global__ void deg_kernel(const int* __restrict__ src, const int* __restrict__ dst,
                           float* __restrict__ dsrc, float* __restrict__ ddst, int E) {
  int e = blockIdx.x * blockDim.x + threadIdx.x;
  if (e < E) {
    atomicAdd(&dsrc[src[e]], 1.0f);
    atomicAdd(&ddst[dst[e]], 1.0f);
  }
}

__global__ void norm_kernel(float* __restrict__ d, int n) {  // deg -> deg^{-1/2}
  int i = blockIdx.x * blockDim.x + threadIdx.x;
  if (i < n) {
    float v = d[i];
    d[i] = (v > 0.0f) ? rsqrtf(v) : 0.0f;
  }
}

// Xb[n,k] = bf16(features[n,k] * c_src[n]),  K = 256
__global__ void prep_x_kernel(const float* __restrict__ x, const float* __restrict__ c,
                              bf16_t* __restrict__ xb, long total, int K) {
  long i = (long)blockIdx.x * blockDim.x + threadIdx.x;
  if (i < total) xb[i] = f2bf(x[i] * c[i / K]);
}

// W1t[n,k] = bf16(W1[k,n]);  W1: [256,128] row-major -> W1t: [128,256]
__global__ void t_w1_kernel(const float* __restrict__ w, bf16_t* __restrict__ wt) {
  int i = blockIdx.x * blockDim.x + threadIdx.x;   // 128*256
  if (i < 128 * 256) {
    int n = i >> 8, k = i & 255;
    wt[i] = f2bf(w[k * 128 + n]);
  }
}

// Wct[n,k] = bf16( n<64 ? Wm[k,n] : Ws[k,n-64] );  -> [128,128] transposed pack
__global__ void t_wc_kernel(const float* __restrict__ wm, const float* __restrict__ ws,
                            bf16_t* __restrict__ wt) {
  int i = blockIdx.x * blockDim.x + threadIdx.x;   // 128*128
  if (i < 128 * 128) {
    int n = i >> 7, k = i & 127;
    float v = (n < 64) ? wm[k * 64 + n] : ws[k * 64 + (n - 64)];
    wt[i] = f2bf(v);
  }
}

// scatter-add over edges: agg[dst[e], :] += T[src[e], :], C columns (C % 4 == 0)
__global__ void scatter_kernel(const float* __restrict__ T, const int* __restrict__ src,
                               const int* __restrict__ dst, float* __restrict__ agg,
                               int E, int C) {
  long t = (long)blockIdx.x * blockDim.x + threadIdx.x;
  int per = C >> 2;
  long e = t / per;
  int c4 = (int)(t % per) << 2;
  if (e >= E) return;
  const float4 m = *(const float4*)(T + (size_t)src[e] * C + c4);
  float* o = agg + (size_t)dst[e] * C + c4;
  atomicAdd(o + 0, m.x);
  atomicAdd(o + 1, m.y);
  atomicAdd(o + 2, m.z);
  atomicAdd(o + 3, m.w);
}

// hb[n,c] = bf16(relu(agg1[n,c]*c_dst[n] + b1[c]) * c_src[n]),  C = 128
__global__ void h_kernel(const float* __restrict__ agg, const float* __restrict__ cdst,
                         const float* __restrict__ csrc, const float* __restrict__ b,
                         bf16_t* __restrict__ hb, long total) {
  long i = (long)blockIdx.x * blockDim.x + threadIdx.x;
  if (i < total) {
    long n = i >> 7;
    int  c = (int)(i & 127);
    float v = agg[i] * cdst[n] + b[c];
    v = fmaxf(v, 0.0f);
    hb[i] = f2bf(v * csrc[n]);
  }
}

// z = (agg2[:, :64]*c_dst + bm) + noise * exp(agg2[:, 64:]*c_dst + bs)
__global__ void z_kernel(const float* __restrict__ agg2, const float* __restrict__ cdst,
                         const float* __restrict__ bm, const float* __restrict__ bs,
                         const float* __restrict__ noise, float* __restrict__ zout,
                         bf16_t* __restrict__ zb, long total) {
  long i = (long)blockIdx.x * blockDim.x + threadIdx.x;
  if (i < total) {
    long n = i >> 6;
    int  c = (int)(i & 63);
    const float* row = agg2 + n * 128;
    float cd = cdst[n];
    float mean = row[c] * cd + bm[c];
    float lstd = row[64 + c] * cd + bs[c];
    float zv = mean + noise[i] * __expf(lstd);
    __builtin_nontemporal_store(zv, zout + i);   // d_out tail, write-once
    zb[i] = f2bf(zv);                            // hot operand, keep cached
  }
}

// ---------------- WMMA GEMM: C[M,Nt] = A[M,K] @ Bt[Nt,K]^T (bf16 -> f32) ----
// one 16x16 tile per wave; block = 8 waves stacked in M (128 rows x 16 cols)
// A row-major, B supplied TRANSPOSED so both fragments are contiguous b128s.

__global__ __launch_bounds__(256)
void gemm_bf16_tn_kernel(const bf16_t* __restrict__ A, const bf16_t* __restrict__ Bt,
                         float* __restrict__ C, int M, int Nt, int K) {
  const int lane = threadIdx.x & 31;
  const int wave = threadIdx.x >> 5;
  const int g    = lane >> 4;     // half-wave group (0/1)
  const int r    = lane & 15;
  const int m0   = blockIdx.x * 128 + wave * 16;
  const int n0   = blockIdx.y * 16;

  v8f acc = {};
  const bf16_t* Arow = A  + (size_t)(m0 + r) * K;   // A: lane=M; K 0-7/16-23 | 8-15/24-31
  const bf16_t* Brow = Bt + (size_t)(n0 + r) * K;   // B: lane=N; K 0-7/16-23 | 8-15/24-31
  for (int k0 = 0; k0 < K; k0 += 32) {
    v8bf alo = *(const v8bf*)(Arow + k0 + g * 8);
    v8bf ahi = *(const v8bf*)(Arow + k0 + 16 + g * 8);
    v8bf blo = *(const v8bf*)(Brow + k0 + g * 8);
    v8bf bhi = *(const v8bf*)(Brow + k0 + 16 + g * 8);
    acc = __builtin_amdgcn_wmma_f32_16x16x32_bf16(false, mk16(alo, ahi),
                                                  false, mk16(blo, bhi),
                                                  (short)0, acc, false, false);
  }
  float* Cp = C + n0 + r;                           // D: vgpr v -> row m0 + g*8 + v
#pragma unroll
  for (int v = 0; v < 8; ++v)
    Cp[(size_t)(m0 + g * 8 + v) * Nt] = acc[v];
}

// ---------------- decoder: pre = sigmoid(z @ z^T), z bf16 [N,64] -----------
// Block = 8 waves covering 128(M) x 64(N). B panel (64 z-rows, 8KB) staged in
// LDS once per block; each wave computes a 16x64 strip: A frags loaded once,
// 4 accumulators, 8 back-to-back v_wmma per wave. Non-temporal stores.

#define DEC_MT 128
#define DEC_NT 64

__global__ __launch_bounds__(256)
void decoder_kernel(const bf16_t* __restrict__ Zb, float* __restrict__ P, int Nn) {
  __shared__ bf16_t sB[DEC_NT * 64];               // 8 KB
  const int tid  = threadIdx.x;
  const int lane = tid & 31;
  const int wave = tid >> 5;
  const int g    = lane >> 4;
  const int r    = lane & 15;
  const int m0   = blockIdx.x * DEC_MT + wave * 16;
  const int n0   = blockIdx.y * DEC_NT;

  // cooperative B-panel load: 8KB = 512 x uint4, 2 per thread
  {
    const uint4* src = (const uint4*)(Zb + (size_t)n0 * 64);
    uint4* dst = (uint4*)sB;
    dst[tid]       = src[tid];
    dst[tid + 256] = src[tid + 256];
  }
  __syncthreads();

  // A fragments for this wave's 16 rows, both k-steps (K = 64)
  const bf16_t* Arow = Zb + (size_t)(m0 + r) * 64;
  v16bf a0 = mk16(*(const v8bf*)(Arow + g * 8),      *(const v8bf*)(Arow + 16 + g * 8));
  v16bf a1 = mk16(*(const v8bf*)(Arow + 32 + g * 8), *(const v8bf*)(Arow + 48 + g * 8));

  v8f acc[4] = {};
#pragma unroll
  for (int nt = 0; nt < 4; ++nt) {
    const bf16_t* Brow = sB + (size_t)(nt * 16 + r) * 64;
    v16bf b0 = mk16(*(const v8bf*)(Brow + g * 8),      *(const v8bf*)(Brow + 16 + g * 8));
    v16bf b1 = mk16(*(const v8bf*)(Brow + 32 + g * 8), *(const v8bf*)(Brow + 48 + g * 8));
    acc[nt] = __builtin_amdgcn_wmma_f32_16x16x32_bf16(false, a0, false, b0,
                                                      (short)0, acc[nt], false, false);
    acc[nt] = __builtin_amdgcn_wmma_f32_16x16x32_bf16(false, a1, false, b1,
                                                      (short)0, acc[nt], false, false);
  }

#pragma unroll
  for (int nt = 0; nt < 4; ++nt) {
#pragma unroll
    for (int v = 0; v < 8; ++v) {
      float s = 1.0f / (1.0f + __expf(-acc[nt][v]));
      // write-once 1.07 GB stream: non-temporal, don't pollute L2
      __builtin_nontemporal_store(
          s, P + (size_t)(m0 + g * 8 + v) * Nn + n0 + nt * 16 + r);
    }
  }
}

// ---------------------------------------------------------------------------

extern "C" void kernel_launch(void* const* d_in, const int* in_sizes, int n_in,
                              void* d_out, int out_size, void* d_ws, size_t ws_size,
                              hipStream_t stream) {
  const float* features = (const float*)d_in[0];   // [N,256]
  const float* W1       = (const float*)d_in[1];   // [256,128]
  const float* b1       = (const float*)d_in[2];   // [128]
  const float* Wm       = (const float*)d_in[3];   // [128,64]
  const float* bm       = (const float*)d_in[4];   // [64]
  const float* Ws       = (const float*)d_in[5];   // [128,64]
  const float* bs       = (const float*)d_in[6];   // [64]
  const float* noise    = (const float*)d_in[7];   // [N,64]
  const int*   esrc     = (const int*)d_in[8];     // [E]
  const int*   edst     = (const int*)d_in[9];     // [E]
  (void)in_sizes; (void)n_in; (void)out_size; (void)ws_size;

  float* pre  = (float*)d_out;                       // [N,N]
  float* zout = (float*)d_out + (size_t)GN * GN;     // [N,64]

  // workspace carve-up (all sizes 256B-aligned)
  char* w = (char*)d_ws;
  float*  deg_src = (float*)w;  w += GN * 4;                  // -> c_src
  float*  deg_dst = (float*)w;  w += GN * 4;                  // -> c_dst
  bf16_t* Xb      = (bf16_t*)w; w += (size_t)GN * 256 * 2;    // scaled features, bf16
  bf16_t* W1t     = (bf16_t*)w; w += 128 * 256 * 2;           // W1 transposed [Nt,K]
  float*  Tbuf    = (float*)w;  w += (size_t)GN * 128 * 4;    // GEMM outputs (reused)
  float*  agg1    = (float*)w;  w += (size_t)GN * 128 * 4;
  bf16_t* Hb      = (bf16_t*)w; w += (size_t)GN * 128 * 2;    // relu(h)*c_src, bf16
  bf16_t* Wct     = (bf16_t*)w; w += 128 * 128 * 2;           // [Wm|Ws] transposed pack
  float*  agg2    = (float*)w;  w += (size_t)GN * 128 * 4;
  bf16_t* Zb      = (bf16_t*)w; w += (size_t)GN * 64 * 2;

  const int B = 256;
  auto blocks = [](long n, int b) { return (unsigned)((n + b - 1) / b); };

  // zero accumulators (every call: deterministic under graph replay)
  zero_f32<<<blocks(2L * GN, B), B, 0, stream>>>(deg_src, 2L * GN);
  zero_f32<<<blocks((long)GN * 128, B), B, 0, stream>>>(agg1, (long)GN * 128);
  zero_f32<<<blocks((long)GN * 128, B), B, 0, stream>>>(agg2, (long)GN * 128);

  // degree norms
  deg_kernel<<<blocks(GE, B), B, 0, stream>>>(esrc, edst, deg_src, deg_dst, GE);
  norm_kernel<<<blocks(2L * GN, B), B, 0, stream>>>(deg_src, 2 * GN);  // contiguous pair

  // layer 1: (x*c_src) @ W1 -> scatter -> relu -> rescale
  prep_x_kernel<<<blocks((long)GN * 256, B), B, 0, stream>>>(features, deg_src, Xb,
                                                             (long)GN * 256, 256);
  t_w1_kernel<<<blocks(128L * 256, B), B, 0, stream>>>(W1, W1t);
  gemm_bf16_tn_kernel<<<dim3(GN / 128, 128 / 16), B, 0, stream>>>(Xb, W1t, Tbuf,
                                                                  GN, 128, 256);
  scatter_kernel<<<blocks((long)GE * (128 / 4), B), B, 0, stream>>>(Tbuf, esrc, edst,
                                                                    agg1, GE, 128);
  h_kernel<<<blocks((long)GN * 128, B), B, 0, stream>>>(agg1, deg_dst, deg_src, b1,
                                                        Hb, (long)GN * 128);

  // layers 2+3 fused: (h*c_src) @ [Wm|Ws] -> scatter -> z
  t_wc_kernel<<<blocks(128L * 128, B), B, 0, stream>>>(Wm, Ws, Wct);
  gemm_bf16_tn_kernel<<<dim3(GN / 128, 128 / 16), B, 0, stream>>>(Hb, Wct, Tbuf,
                                                                  GN, 128, 128);
  scatter_kernel<<<blocks((long)GE * (128 / 4), B), B, 0, stream>>>(Tbuf, esrc, edst,
                                                                    agg2, GE, 128);
  z_kernel<<<blocks((long)GN * 64, B), B, 0, stream>>>(agg2, deg_dst, bm, bs, noise,
                                                       zout, Zb, (long)GN * 64);

  // decoder: sigmoid(z @ z^T) — store-bound; LDS-staged B panel, 8 wmma/wave
  decoder_kernel<<<dim3(GN / DEC_MT, GN / DEC_NT), B, 0, stream>>>(Zb, pre, GN);
}
